// GroupQueryAttention_17592186044427
// MI455X (gfx1250) — matrix operand order repaired and checked by reading, not compiled, not run on hardware
//
#include <hip/hip_runtime.h>
#include <hip/hip_bf16.h>

typedef __attribute__((ext_vector_type(16))) _Float16 v16h;
typedef __attribute__((ext_vector_type(8)))  float    v8f;
typedef __attribute__((__vector_size__(4 * sizeof(int)))) int v4i;

#define HID   2048
#define SEQN  2048
#define KVD   512
#define NHEAD 32
#define NGRP  8
#define HDIM  64

#define AS1 __attribute__((address_space(1)))
#define AS3 __attribute__((address_space(3)))

// ---- CDNA5 async global->LDS path (guarded: falls back to direct loads) ----
#if defined(__has_builtin)
#if __has_builtin(__builtin_amdgcn_global_load_async_to_lds_b128)
#define ASYNC_LDS 1
#endif
#endif
#ifndef ASYNC_LDS
#define ASYNC_LDS 0
#endif

static __device__ inline void async_copy_b128(const _Float16* gp, _Float16* lp) {
#if ASYNC_LDS
  __builtin_amdgcn_global_load_async_to_lds_b128(
      (AS1 v4i*)(unsigned long long)(uintptr_t)gp,
      (AS3 v4i*)(unsigned)(uintptr_t)lp, 0, 0);
#else
  (void)gp; (void)lp;
#endif
}

template <int N>
static __device__ inline void wait_asynccnt() {
#if defined(__has_builtin) && __has_builtin(__builtin_amdgcn_s_wait_asynccnt)
  __builtin_amdgcn_s_wait_asynccnt((short)N);
#else
  asm volatile("s_wait_asynccnt %0" ::"i"(N) : "memory");
#endif
}

union ABFrag { uint4 q[2]; v16h v; };

static __device__ inline v8f wmma_f16(const v16h& a, const v16h& b, const v8f& c) {
  return __builtin_amdgcn_wmma_f32_16x16x32_f16(false, a, false, b, (short)0, c, false, false);
}

// ---------------- stage 1: f32 -> f16 convert ----------------
__global__ void f32_to_f16_kernel(const float* __restrict__ a, _Float16* __restrict__ o, int n) {
  int i = blockIdx.x * blockDim.x + threadIdx.x;
  if (i < n) o[i] = (_Float16)a[i];
}

// ---------------- stage 2: W (KxN f32) -> Wt (NxK f16), LDS-tiled ----------------
__global__ __launch_bounds__(256)
void w_transpose_f16_kernel(const float* __restrict__ W, _Float16* __restrict__ Wt,
                            int K, int N) {
  __shared__ _Float16 tile[32][33];
  const int n0 = blockIdx.x * 32, k0 = blockIdx.y * 32;
  const int tx = threadIdx.x & 31, ty = threadIdx.x >> 5;   // 32 x 8 threads
#pragma unroll
  for (int i = 0; i < 4; ++i)
    tile[ty + i * 8][tx] = (_Float16)W[(size_t)(k0 + ty + i * 8) * N + n0 + tx];
  __syncthreads();
#pragma unroll
  for (int i = 0; i < 4; ++i)
    Wt[(size_t)(n0 + ty + i * 8) * K + k0 + tx] = tile[tx][ty + i * 8];
}

// ---------------- stage 4: f16 transpose (V -> Vt), LDS-tiled ----------------
__global__ __launch_bounds__(256)
void f16_transpose_kernel(const _Float16* __restrict__ S, _Float16* __restrict__ D,
                          int R, int C) {   // S[R,C] -> D[C,R]
  __shared__ _Float16 tile[32][33];
  const int c0 = blockIdx.x * 32, r0 = blockIdx.y * 32;
  const int tx = threadIdx.x & 31, ty = threadIdx.x >> 5;
#pragma unroll
  for (int i = 0; i < 4; ++i)
    tile[ty + i * 8][tx] = S[(size_t)(r0 + ty + i * 8) * C + c0 + tx];
  __syncthreads();
#pragma unroll
  for (int i = 0; i < 4; ++i)
    D[(size_t)(c0 + ty + i * 8) * R + r0 + tx] = tile[tx][ty + i * 8];
}

// ---------------- stage 3/6: WMMA GEMM ----------------
// C[M,N] = A[M,K](f16 rm) x Bt[N,K](f16, transposed weights) + bias
// one wave -> 32x64 output tile: 2 A-frags x 4 B-frags -> 8 accumulators.
// B-frag reused 2x, A-frag 4x: ~21 FLOP/byte of (L2-resident) loads.
template <bool OUT_F32>
__global__ __launch_bounds__(256)
void gemm_wmma_kernel(const _Float16* __restrict__ A, const _Float16* __restrict__ Bt,
                      const float* __restrict__ bias, void* __restrict__ Cout,
                      int M, int N, int K) {
  const int wid  = (blockIdx.x * blockDim.x + threadIdx.x) >> 5;
  const int lane = threadIdx.x & 31;
  const int half = lane >> 4, l15 = lane & 15;
  const int tiles_n = N >> 6;
  const int tm = wid / tiles_n, tn = wid - tm * tiles_n;   // tm over M/32
  if (tm * 32 >= M) return;

  const _Float16* arow0 = A + (size_t)(tm * 32 + l15) * K;
  const _Float16* arow1 = arow0 + (size_t)16 * K;
  const _Float16* brow[4];
#pragma unroll
  for (int j = 0; j < 4; ++j)
    brow[j] = Bt + (size_t)(tn * 64 + j * 16 + l15) * K;

  v8f acc0[4] = {v8f{}, v8f{}, v8f{}, v8f{}};
  v8f acc1[4] = {v8f{}, v8f{}, v8f{}, v8f{}};

  for (int kk = 0; kk < K; kk += 32) {
    ABFrag a0, a1;
    a0.q[0] = *(const uint4*)(arow0 + kk + half * 8);
    a0.q[1] = *(const uint4*)(arow0 + kk + 16 + half * 8);
    a1.q[0] = *(const uint4*)(arow1 + kk + half * 8);
    a1.q[1] = *(const uint4*)(arow1 + kk + 16 + half * 8);
    __builtin_prefetch(arow0 + kk + 64, 0, 0);
    __builtin_prefetch(arow1 + kk + 64, 0, 0);
#pragma unroll
    for (int j = 0; j < 4; ++j) {
      ABFrag b;
      const _Float16* bp = brow[j] + kk + half * 8;
      b.q[0] = *(const uint4*)(bp);
      b.q[1] = *(const uint4*)(bp + 16);
      acc0[j] = wmma_f16(a0.v, b.v, acc0[j]);
      acc1[j] = wmma_f16(a1.v, b.v, acc1[j]);
    }
  }

#pragma unroll
  for (int j = 0; j < 4; ++j) {
    const int n = tn * 64 + j * 16 + l15;
    const float bv = bias ? bias[n] : 0.0f;
#pragma unroll
    for (int r = 0; r < 8; ++r) {
      const int m0 = tm * 32 + r + 8 * half;          // C layout: VGPR r -> row r+8*half
      const float v0 = acc0[j][r] + bv;
      const float v1 = acc1[j][r] + bv;
      if (OUT_F32) {
        ((float*)Cout)[(size_t)m0 * N + n]        = v0;
        ((float*)Cout)[(size_t)(m0 + 16) * N + n] = v1;
      } else {
        ((_Float16*)Cout)[(size_t)m0 * N + n]        = (_Float16)v0;
        ((_Float16*)Cout)[(size_t)(m0 + 16) * N + n] = (_Float16)v1;
      }
    }
  }
}

// ---------------- stage 5: flash GQA attention ----------------
// 4 waves/block, one wave per (head, 16-query block); K tiles of 32 keys.
// K tiles are double-buffered in LDS via async global->LDS DMA when available.
__global__ __launch_bounds__(128)
void gqa_attention_kernel(const _Float16* __restrict__ Qh, const _Float16* __restrict__ Kh,
                          const _Float16* __restrict__ Vt, _Float16* __restrict__ Oh) {
  const int lane = threadIdx.x & 31;
  const int w    = threadIdx.x >> 5;            // 0..3
  const int h    = blockIdx.x;                  // head
  const int qb   = blockIdx.y * 4 + w;          // 16-row query block
  const int g    = h >> 2;                      // KV group (4 heads/group)
  const int half = lane >> 4, l15 = lane & 15;

  __shared__ __align__(16) _Float16 plds[4][16 * 40];     // per-wave P tile
#if ASYNC_LDS
  __shared__ __align__(16) _Float16 kst[4][2][32 * 64];   // per-wave double-buffered K tile
#endif

  // Q fragments: rows qb*16..+15, head dim in two 32-wide K-chunks
  const _Float16* qbase = Qh + (size_t)(qb * 16 + l15) * HID + h * HDIM;
  ABFrag qa0, qa1;
  qa0.q[0] = *(const uint4*)(qbase + 0  + half * 8);
  qa0.q[1] = *(const uint4*)(qbase + 16 + half * 8);
  qa1.q[0] = *(const uint4*)(qbase + 32 + half * 8);
  qa1.q[1] = *(const uint4*)(qbase + 48 + half * 8);

  v8f o0 = {}, o1 = {}, o2 = {}, o3 = {};
  float mrow[8], lrow[8];
#pragma unroll
  for (int r = 0; r < 8; ++r) { mrow[r] = -1e30f; lrow[r] = 0.0f; }

  const int mbase = qb * 16 + 8 * half;         // global row base for C-frag rows
  const int ntile = (qb * 16 + 47) >> 5;        // ceil((qb*16+16)/32) causal key tiles

#if ASYNC_LDS
  // async-stage one 32x64 K tile (4 KB): 8 x wave-wide b128 transfers
  const int srow = lane >> 3, scol = (lane & 7) * 8;    // 4 rows per issue, 16B per lane
  {
    const _Float16* gb = Kh + (size_t)0 * KVD + g * HDIM;
#pragma unroll
    for (int e = 0; e < 8; ++e)
      async_copy_b128(gb + (size_t)(e * 4 + srow) * KVD + scol,
                      &kst[w][0][(e * 4 + srow) * 64 + scol]);
  }
#endif

  for (int t = 0; t < ntile; ++t) {
    const int t0 = t * 32;
    v8f s0 = {}, s1 = {};

#if ASYNC_LDS
    if (t + 1 < ntile) {                         // prefetch next tile into other buffer
      const _Float16* gb = Kh + (size_t)(t0 + 32) * KVD + g * HDIM;
#pragma unroll
      for (int e = 0; e < 8; ++e)
        async_copy_b128(gb + (size_t)(e * 4 + srow) * KVD + scol,
                        &kst[w][(t + 1) & 1][(e * 4 + srow) * 64 + scol]);
      wait_asynccnt<8>();                        // current tile's 8 transfers done
    } else {
      wait_asynccnt<0>();
    }
    const _Float16* kb = kst[w][t & 1];
    {
      const _Float16* kr0 = kb + l15 * 64;              // key t0+l15
      const _Float16* kr1 = kb + (16 + l15) * 64;       // key t0+16+l15
      ABFrag b;
      b.q[0] = *(const uint4*)(kr0 + half * 8);      b.q[1] = *(const uint4*)(kr0 + half * 8 + 16);
      s0 = wmma_f16(qa0.v, b.v, s0);
      b.q[0] = *(const uint4*)(kr0 + 32 + half * 8); b.q[1] = *(const uint4*)(kr0 + 32 + half * 8 + 16);
      s0 = wmma_f16(qa1.v, b.v, s0);
      b.q[0] = *(const uint4*)(kr1 + half * 8);      b.q[1] = *(const uint4*)(kr1 + half * 8 + 16);
      s1 = wmma_f16(qa0.v, b.v, s1);
      b.q[0] = *(const uint4*)(kr1 + 32 + half * 8); b.q[1] = *(const uint4*)(kr1 + 32 + half * 8 + 16);
      s1 = wmma_f16(qa1.v, b.v, s1);
    }
#else
    {
      const _Float16* kr0 = Kh + (size_t)(t0 + l15) * KVD + g * HDIM;
      const _Float16* kr1 = Kh + (size_t)(t0 + 16 + l15) * KVD + g * HDIM;
      ABFrag b;
      b.q[0] = *(const uint4*)(kr0 + half * 8);      b.q[1] = *(const uint4*)(kr0 + half * 8 + 16);
      s0 = wmma_f16(qa0.v, b.v, s0);
      b.q[0] = *(const uint4*)(kr0 + 32 + half * 8); b.q[1] = *(const uint4*)(kr0 + 32 + half * 8 + 16);
      s0 = wmma_f16(qa1.v, b.v, s0);
      b.q[0] = *(const uint4*)(kr1 + half * 8);      b.q[1] = *(const uint4*)(kr1 + half * 8 + 16);
      s1 = wmma_f16(qa0.v, b.v, s1);
      b.q[0] = *(const uint4*)(kr1 + 32 + half * 8); b.q[1] = *(const uint4*)(kr1 + 32 + half * 8 + 16);
      s1 = wmma_f16(qa1.v, b.v, s1);
    }
#endif

    // ---- scale + causal mask (col = t0[+16]+l15, row = mbase+r)
    const float scl = 0.125f;                    // 1/sqrt(64)
#pragma unroll
    for (int r = 0; r < 8; ++r) {
      float v0 = s0[r] * scl; if (t0 + l15      > mbase + r) v0 = -1e9f;
      float v1 = s1[r] * scl; if (t0 + 16 + l15 > mbase + r) v1 = -1e9f;
      s0[r] = v0; s1[r] = v1;
    }
    // ---- online softmax
    float escale[8];
#pragma unroll
    for (int r = 0; r < 8; ++r) {
      float mx = fmaxf(s0[r], s1[r]);
      mx = fmaxf(mx, __shfl_xor(mx, 1, 32));
      mx = fmaxf(mx, __shfl_xor(mx, 2, 32));
      mx = fmaxf(mx, __shfl_xor(mx, 4, 32));
      mx = fmaxf(mx, __shfl_xor(mx, 8, 32));
      const float nm = fmaxf(mrow[r], mx);
      escale[r] = __expf(mrow[r] - nm);
      mrow[r] = nm;
    }
    _Float16* pw = plds[w];
#pragma unroll
    for (int r = 0; r < 8; ++r) {
      const float p0 = __expf(s0[r] - mrow[r]);
      const float p1 = __expf(s1[r] - mrow[r]);
      float sum = p0 + p1;
      sum += __shfl_xor(sum, 1, 32);
      sum += __shfl_xor(sum, 2, 32);
      sum += __shfl_xor(sum, 4, 32);
      sum += __shfl_xor(sum, 8, 32);
      lrow[r] = lrow[r] * escale[r] + sum;
      o0[r] *= escale[r]; o1[r] *= escale[r]; o2[r] *= escale[r]; o3[r] *= escale[r];
      const int mr = r + 8 * half;
      pw[mr * 40 + l15]      = (_Float16)p0;
      pw[mr * 40 + 16 + l15] = (_Float16)p1;
    }
    asm volatile("s_wait_dscnt 0x0" ::: "memory");
    // ---- reload P as A-fragment
    ABFrag pa;
    pa.q[0] = *(const uint4*)(pw + l15 * 40 + half * 8);
    pa.q[1] = *(const uint4*)(pw + l15 * 40 + 16 + half * 8);
    // ---- O += P @ V  (B-frag lane = output dim, contiguous keys from Vt rows)
    {
      const _Float16* vr0 = Vt + (size_t)(g * HDIM + 0  + l15) * SEQN + t0 + half * 8;
      const _Float16* vr1 = Vt + (size_t)(g * HDIM + 16 + l15) * SEQN + t0 + half * 8;
      const _Float16* vr2 = Vt + (size_t)(g * HDIM + 32 + l15) * SEQN + t0 + half * 8;
      const _Float16* vr3 = Vt + (size_t)(g * HDIM + 48 + l15) * SEQN + t0 + half * 8;
      ABFrag vb;
      vb.q[0] = *(const uint4*)vr0; vb.q[1] = *(const uint4*)(vr0 + 16);
      o0 = wmma_f16(pa.v, vb.v, o0);
      vb.q[0] = *(const uint4*)vr1; vb.q[1] = *(const uint4*)(vr1 + 16);
      o1 = wmma_f16(pa.v, vb.v, o1);
      vb.q[0] = *(const uint4*)vr2; vb.q[1] = *(const uint4*)(vr2 + 16);
      o2 = wmma_f16(pa.v, vb.v, o2);
      vb.q[0] = *(const uint4*)vr3; vb.q[1] = *(const uint4*)(vr3 + 16);
      o3 = wmma_f16(pa.v, vb.v, o3);
    }
  }
  // ---- normalize + store O tile (16 x 64)
  float inv[8];
#pragma unroll
  for (int r = 0; r < 8; ++r) inv[r] = 1.0f / lrow[r];
#pragma unroll
  for (int r = 0; r < 8; ++r) {
    _Float16* orow = Oh + (size_t)(qb * 16 + r + 8 * half) * HID + h * HDIM + l15;
    orow[0]  = (_Float16)(o0[r] * inv[r]);
    orow[16] = (_Float16)(o1[r] * inv[r]);
    orow[32] = (_Float16)(o2[r] * inv[r]);
    orow[48] = (_Float16)(o3[r] * inv[r]);
  }
}

// ---------------- host: stage launcher ----------------
extern "C" void kernel_launch(void* const* d_in, const int* in_sizes, int n_in,
                              void* d_out, int out_size, void* d_ws, size_t ws_size,
                              hipStream_t stream) {
  const float* x  = (const float*)d_in[0];
  // d_in[1] = causal_mask (applied analytically)
  const float* Wq = (const float*)d_in[2];
  const float* bq = (const float*)d_in[3];
  const float* Wk = (const float*)d_in[4];
  const float* bk = (const float*)d_in[5];
  const float* Wv = (const float*)d_in[6];
  const float* bv = (const float*)d_in[7];
  const float* Wo = (const float*)d_in[8];
  const float* bo = (const float*)d_in[9];
  float* out = (float*)d_out;

  char* ws = (char*)d_ws;
  const size_t MB = 1024 * 1024;
  _Float16* xh  = (_Float16*)(ws + 0);          //  8 MB  x f16       [2048,2048]
  _Float16* Wqt = (_Float16*)(ws + 8  * MB);    //  8 MB  Wq^T        [2048,2048]
  _Float16* Wkt = (_Float16*)(ws + 16 * MB);    //  2 MB  Wk^T        [512,2048]
  _Float16* Wvt = (_Float16*)(ws + 18 * MB);    //  2 MB  Wv^T        [512,2048]
  _Float16* Wot = (_Float16*)(ws + 20 * MB);    //  8 MB  Wo^T        [2048,2048]
  _Float16* Qh  = (_Float16*)(ws + 28 * MB);    //  8 MB  Q           [2048,2048]
  _Float16* Kh  = (_Float16*)(ws + 36 * MB);    //  2 MB  K           [2048,512]
  _Float16* Vh  = (_Float16*)(ws + 38 * MB);    //  2 MB  V           [2048,512]
  _Float16* Vt  = (_Float16*)(ws + 40 * MB);    //  2 MB  V^T         [512,2048]
  _Float16* Oh  = (_Float16*)(ws + 42 * MB);    //  8 MB  attn out    [2048,2048]

  const int TB = 256;
  // stage 1/2: conversions (LDS-tiled transposes, coalesced both sides)
  f32_to_f16_kernel<<<(HID * SEQN + TB - 1) / TB, TB, 0, stream>>>(x, xh, HID * SEQN);
  w_transpose_f16_kernel<<<dim3(HID / 32, HID / 32), TB, 0, stream>>>(Wq, Wqt, HID, HID);
  w_transpose_f16_kernel<<<dim3(KVD / 32, HID / 32), TB, 0, stream>>>(Wk, Wkt, HID, KVD);
  w_transpose_f16_kernel<<<dim3(KVD / 32, HID / 32), TB, 0, stream>>>(Wv, Wvt, HID, KVD);
  w_transpose_f16_kernel<<<dim3(HID / 32, HID / 32), TB, 0, stream>>>(Wo, Wot, HID, HID);

  // stage 3: projections (waves = (M/32)*(N/64), 8 waves/block)
  gemm_wmma_kernel<false><<<(SEQN / 32) * (HID / 64) / 8, TB, 0, stream>>>(xh, Wqt, bq, Qh, SEQN, HID, HID);
  gemm_wmma_kernel<false><<<(SEQN / 32) * (KVD / 64) / 8, TB, 0, stream>>>(xh, Wkt, bk, Kh, SEQN, KVD, HID);
  gemm_wmma_kernel<false><<<(SEQN / 32) * (KVD / 64) / 8, TB, 0, stream>>>(xh, Wvt, bv, Vh, SEQN, KVD, HID);

  // stage 4: V transpose for PV WMMA B-fragments
  f16_transpose_kernel<<<dim3(KVD / 32, SEQN / 32), TB, 0, stream>>>(Vh, Vt, SEQN, KVD);

  // stage 5: flash GQA attention (32 heads x 32 blocks of 4 query-block waves)
  gqa_attention_kernel<<<dim3(NHEAD, SEQN / 16 / 4), 128, 0, stream>>>(Qh, Kh, Vt, Oh);

  // stage 6: output projection -> f32
  gemm_wmma_kernel<true><<<(SEQN / 32) * (HID / 64) / 8, TB, 0, stream>>>(Oh, Wot, bo, out, SEQN, HID, HID);
}
